// LocalPoolPointnet_9912784519391
// MI455X (gfx1250) — matrix operand order repaired and checked by reading, not compiled, not run on hardware
//
#include <hip/hip_runtime.h>
#include <hip/hip_bf16.h>

// ---------------- types / constants ----------------
typedef __attribute__((ext_vector_type(16))) __bf16 v16bf;
typedef __attribute__((ext_vector_type(8)))  float  v8f;
typedef __attribute__((ext_vector_type(4)))  int    v4i;

#define RESO   64
#define R2V    4096
#define NBATCH 16

struct u32x8 { uint4 lo, hi; };   // 32B carrier for a WMMA bf16 fragment

// Device-side availability of the gfx1250 async global->LDS copy path.
// (Host pass sees 0; only the device function BODY differs, signatures match.)
#if defined(__HIP_DEVICE_COMPILE__) && __has_builtin(__builtin_amdgcn_global_load_async_to_lds_b128)
#define HAVE_ASYNC_LDS 1
#else
#define HAVE_ASYNC_LDS 0
#endif

// ---------------- helpers ----------------
__device__ __forceinline__ unsigned short f2bf(float f) {
  unsigned int x = __builtin_bit_cast(unsigned int, f);
  x += 0x7FFFu + ((x >> 16) & 1u);            // round-to-nearest-even
  return (unsigned short)(x >> 16);
}
__device__ __forceinline__ float bf2f(unsigned short h) {
  return __builtin_bit_cast(float, (unsigned int)h << 16);
}
// branchless packed-bf16 ReLU: zero each 16-bit half whose sign bit is set
__device__ __forceinline__ unsigned int relu2(unsigned int u) {
  unsigned int s = u & 0x80008000u;
  return u & ~(s | (s - (s >> 15)));
}
// order-preserving uint encoding of float (monotone: a<b  <=>  enc(a)<enc(b))
__device__ __forceinline__ unsigned int ordenc(float f) {
  unsigned int u = __builtin_bit_cast(unsigned int, f);
  return (u & 0x80000000u) ? ~u : (u | 0x80000000u);
}
__device__ __forceinline__ float orddec(unsigned int u) {
  unsigned int b = (u & 0x80000000u) ? (u & 0x7FFFFFFFu) : ~u;
  return __builtin_bit_cast(float, b);
}

#if HAVE_ASYNC_LDS
// builtin signature (from clang diagnostic): (int4 AS1*, int4 AS3*, int, int)
typedef __attribute__((address_space(1))) v4i gv4i;
typedef __attribute__((address_space(3))) v4i lv4i;
__device__ __forceinline__ void async_g2l_b128(const void* g, void* l) {
  __builtin_amdgcn_global_load_async_to_lds_b128((gv4i*)g, (lv4i*)l, 0, 0);
}
__device__ __forceinline__ void wait_async0() {
#if __has_builtin(__builtin_amdgcn_s_wait_asynccnt)
  __builtin_amdgcn_s_wait_asynccnt(0);
#else
  asm volatile("s_wait_asynccnt 0" ::: "memory");
#endif
}
#endif

// ---------------- plane indices ----------------
__global__ void k_idx(const float* __restrict__ p, int* __restrict__ idx, int N) {
  int n = blockIdx.x * blockDim.x + threadIdx.x;
  if (n >= N) return;
  const float inv = 1.0f / 1.101f;   // 1/(1+PAD+1e-3)
  float q0 = fminf(fmaxf(p[3 * n + 0] * inv + 0.5f, 0.0f), 1.0f - 1e-6f);
  float q1 = fminf(fmaxf(p[3 * n + 1] * inv + 0.5f, 0.0f), 1.0f - 1e-6f);
  float q2 = fminf(fmaxf(p[3 * n + 2] * inv + 0.5f, 0.0f), 1.0f - 1e-6f);
  int i0 = min((int)(q0 * (float)RESO), RESO - 1);
  int i1 = min((int)(q1 * (float)RESO), RESO - 1);
  int i2 = min((int)(q2 * (float)RESO), RESO - 1);
  idx[0 * N + n] = i0 + RESO * i2;   // 'xz' dims (0,2)
  idx[1 * N + n] = i0 + RESO * i1;   // 'xy' dims (0,1)
  idx[2 * N + n] = i1 + RESO * i2;   // 'yz' dims (1,2)
}

// ---------------- fc_pos: (N,3)@(3,256)+b -> netcat bf16 (N,256) ----------------
__global__ void k_fcpos(const float* __restrict__ p, const float* __restrict__ w,
                        const float* __restrict__ b, unsigned short* __restrict__ netcat) {
  long e = (long)blockIdx.x * 256 + threadIdx.x;  // e = n*256 + j
  int n = (int)(e >> 8);
  int j = (int)(e & 255);
  float v = p[3 * n] * w[j] + p[3 * n + 1] * w[256 + j] + p[3 * n + 2] * w[512 + j] + b[j];
  netcat[e] = f2bf(v);
}

// ---------------- generic bf16 WMMA GEMM ----------------
// out[row,col] = (RELU ? relu(A) : A)[row,0:K] @ bf16(W[0:K,0:128])
//                (+ bias[col] if BIAS) (+ addC[row,col] if ADDC)
// A: bf16 (ushort), row stride lda. W: f32 (K,128) row-major. out: bf16, stride ldo.
// Block: 256 threads (8 waves). Block tile: 128 rows x 128 cols. K in {128, 256}.
//
// K-permutation trick: A is staged in natural memory order (pure contiguous row
// copy -> async b128 capable). The hardware A-fragment (2 x ds_load_b128 of
// slots [hi*8, hi*8+8)) then sees K order pi = [0-7,16-23 | 8-15,24-31] mapped
// to memory K [0-15 | 16-31]. GEMM is invariant if B uses the same permutation,
// so B staging gathers W rows in the same contiguous k-pair order (identity, no swap).
template <int K, bool RELU, bool BIAS, bool ADDC>
__global__ __launch_bounds__(256) void k_gemm(
    const unsigned short* __restrict__ A, int lda,
    const float* __restrict__ W, const float* __restrict__ bias,
    const unsigned short* __restrict__ addC, int ldadd,
    unsigned short* __restrict__ out, int ldo) {
  // Row stride 20 dwords (80B): 16B aligned for b128, bank-tiling for frags.
  __shared__ __align__(16) unsigned int Asp[128][20];  // [row][kpair 0..15]
  __shared__ __align__(16) unsigned int Bsp[128][20];  // [col][hw kpair 0..15]

  const int t     = threadIdx.x;
  const int lane  = t & 31;
  const int wave  = t >> 5;
  const int halfl = lane & 15;
  const int hi    = lane >> 4;
  const size_t rowBase = (size_t)blockIdx.x * 128;

  v8f acc[8];
#pragma unroll
  for (int i = 0; i < 8; ++i)
#pragma unroll
    for (int j = 0; j < 8; ++j) acc[i][j] = 0.0f;

#pragma unroll
  for (int k0 = 0; k0 < K; k0 += 32) {
    // ---- stage A tile: 128 rows x 32 k, natural order (contiguous rows) ----
#if HAVE_ASYNC_LDS
    if constexpr (!RELU) {
      // pure copy: async global->LDS, 16B per lane-op
#pragma unroll
      for (int j = 0; j < 2; ++j) {
        int e = t + 256 * j;     // [0,512): 128 rows x 4 x b128
        int r = e >> 2;
        int c = e & 3;
        async_g2l_b128(A + (rowBase + (size_t)r) * (size_t)lda + k0 + 8 * c,
                       &Asp[r][4 * c]);
      }
      wait_async0();
    } else
#endif
    {
#pragma unroll
      for (int j = 0; j < 4; ++j) {
        int e  = t + 256 * j;    // [0,1024): 128 rows x 8 uint2
        int r  = e >> 3;
        int kq = e & 7;
        const unsigned short* src =
            A + (rowBase + (size_t)r) * (size_t)lda + k0 + 4 * kq;
        uint2 u = *(const uint2*)src;
        if constexpr (RELU) {
          u.x = relu2(u.x);
          u.y = relu2(u.y);
        }
        if (k0 + 32 < K) __builtin_prefetch(src + 32, 0, 0);
        *(uint2*)&Asp[r][2 * kq] = u;
      }
    }
    // ---- stage W tile transposed in the same contiguous k-pair order as A ----
#pragma unroll
    for (int j = 0; j < 8; ++j) {
      int e   = t + 256 * j;     // [0,2048): 16 kpairs x 128 cols
      int kp  = e >> 7;          // hardware k-pair
      int col = e & 127;
      int q   = kp >> 2;         // memory k-pair: identity (contiguous, matches A)
      int mkp = kp;
      float w0 = W[(size_t)(k0 + 2 * mkp) * 128 + col];
      float w1 = W[(size_t)(k0 + 2 * mkp + 1) * 128 + col];
      Bsp[col][kp] = (unsigned int)f2bf(w0) | ((unsigned int)f2bf(w1) << 16);
    }
    __syncthreads();

    // A fragment: 8 contiguous dwords -> two ds_load_b128, no shuffles
    const int M = wave * 16 + halfl;
    u32x8 araw;
    araw.lo = *(const uint4*)&Asp[M][hi * 8];
    araw.hi = *(const uint4*)&Asp[M][hi * 8 + 4];
    v16bf afrag = __builtin_bit_cast(v16bf, araw);

    // 8 N-tiles of B, one WMMA each
#pragma unroll
    for (int nt = 0; nt < 8; ++nt) {
      const int col = nt * 16 + halfl;
      u32x8 braw;
      braw.lo = *(const uint4*)&Bsp[col][hi * 8];
      braw.hi = *(const uint4*)&Bsp[col][hi * 8 + 4];
      v16bf bfrag = __builtin_bit_cast(v16bf, braw);
      acc[nt] = __builtin_amdgcn_wmma_f32_16x16x32_bf16(
          false, afrag, false, bfrag, (short)0, acc[nt], false, false);
    }
    __syncthreads();
  }

  // store: C layout -> VGPR r = row (r + 8*hi) within tile, col = halfl in N-tile
#pragma unroll
  for (int nt = 0; nt < 8; ++nt) {
    const int col = nt * 16 + halfl;
    float bv = 0.0f;
    if constexpr (BIAS) bv = bias[col];
#pragma unroll
    for (int r = 0; r < 8; ++r) {
      size_t row = rowBase + (size_t)(wave * 16 + r + 8 * hi);
      float v = acc[nt][r] + bv;
      if constexpr (ADDC) v += bf2f(addC[row * (size_t)ldadd + col]);
      out[row * (size_t)ldo + col] = f2bf(v);
    }
  }
}

// ---------------- pooling: scatter-max (ordered-uint atomicMax) ----------------
__global__ void k_scatter_max(const unsigned short* __restrict__ netcat,
                              const int* __restrict__ idx,
                              unsigned int* __restrict__ bins, int N, int T) {
  long g = (long)blockIdx.x * 256 + threadIdx.x;  // g = n*128 + c
  int n = (int)(g >> 7);
  int c = (int)(g & 127);
  int b = n / T;
  unsigned int u = ordenc(bf2f(netcat[(size_t)n * 256 + c]));
#pragma unroll
  for (int k = 0; k < 3; ++k) {
    int r2 = idx[(size_t)k * N + n];
    atomicMax(&bins[(((size_t)(k * NBATCH + b)) * R2V + r2) * 128 + c], u);
  }
}

// gather: pooled[n,c] = sum_k bins[k, b, idx_k[n], c] -> netcat cols 128..255
__global__ void k_gather(unsigned short* __restrict__ netcat,
                         const int* __restrict__ idx,
                         const unsigned int* __restrict__ bins, int N, int T) {
  long g = (long)blockIdx.x * 256 + threadIdx.x;
  int n = (int)(g >> 7);
  int c = (int)(g & 127);
  int b = n / T;
  float s = 0.0f;
#pragma unroll
  for (int k = 0; k < 3; ++k) {
    int r2 = idx[(size_t)k * N + n];
    s += orddec(bins[(((size_t)(k * NBATCH + b)) * R2V + r2) * 128 + c]);
  }
  netcat[(size_t)n * 256 + 128 + c] = f2bf(s);
}

// ---------------- final scatter-mean into output plane layout ----------------
__global__ void k_scatter_mean(const unsigned short* __restrict__ cbuf,
                               const int* __restrict__ idx,
                               float* __restrict__ out, float* __restrict__ cnt,
                               int N, int T) {
  long g = (long)blockIdx.x * 256 + threadIdx.x;
  int n = (int)(g >> 7);
  int c = (int)(g & 127);
  int b = n / T;
  float v = bf2f(cbuf[(size_t)n * 128 + c]);
#pragma unroll
  for (int k = 0; k < 3; ++k) {
    int r2 = idx[(size_t)k * N + n];
    atomicAdd(&out[(((size_t)(k * NBATCH + b)) * 128 + c) * R2V + r2], v);
    if (c == 0) atomicAdd(&cnt[((size_t)(k * NBATCH + b)) * R2V + r2], 1.0f);
  }
}

__global__ void k_div(float* __restrict__ out, const float* __restrict__ cnt, long total) {
  long e = (long)blockIdx.x * 256 + threadIdx.x;
  if (e >= total) return;
  long q  = e / (128L * R2V);   // k*NBATCH + b
  long r2 = e & (R2V - 1);
  out[e] /= fmaxf(cnt[q * R2V + r2], 1.0f);
}

// ---------------- host ----------------
extern "C" void kernel_launch(void* const* d_in, const int* in_sizes, int n_in,
                              void* d_out, int out_size, void* d_ws, size_t ws_size,
                              hipStream_t stream) {
  const float* p        = (const float*)d_in[0];
  const float* fc_pos_w = (const float*)d_in[1];
  const float* fc_pos_b = (const float*)d_in[2];
  const float* blk0_w   = (const float*)d_in[3];
  const float* blk0_b   = (const float*)d_in[4];
  const float* blk1_w   = (const float*)d_in[5];
  const float* blk1_b   = (const float*)d_in[6];
  const float* blk_sc_w = (const float*)d_in[7];
  const float* fc_c_w   = (const float*)d_in[8];
  const float* fc_c_b   = (const float*)d_in[9];
  float* out = (float*)d_out;

  const int N = in_sizes[0] / 3;     // B*T = 262144
  const int T = N / NBATCH;

  // workspace layout
  char* ws = (char*)d_ws;
  size_t off = 0;
  int* idxb = (int*)(ws + off);                 off += (size_t)3 * N * 4;
  unsigned short* netcat = (unsigned short*)(ws + off); off += (size_t)N * 256 * 2;
  unsigned short* hbuf   = (unsigned short*)(ws + off); off += (size_t)N * 128 * 2;
  unsigned int*  bins = (unsigned int*)(ws + off);      // overlaid with tbuf
  unsigned short* tbuf = (unsigned short*)(ws + off);
  const size_t binsBytes = (size_t)3 * NBATCH * R2V * 128 * 4;
  off += binsBytes;
  float* cnt = (float*)(ws + off);              off += (size_t)3 * NBATCH * R2V * 4;
  (void)ws_size; (void)n_in;

  const int gemmBlocks = N / 128;

  // 1. plane indices
  k_idx<<<(N + 255) / 256, 256, 0, stream>>>(p, idxb, N);

  // 2. fc_pos -> netcat (all 256 cols = block-0 input)
  k_fcpos<<<N, 256, 0, stream>>>(p, fc_pos_w, fc_pos_b, netcat);

  // 3. 5 ResNet blocks with pooling before blocks 1..4
  for (int i = 0; i < 5; ++i) {
    if (i > 0) {
      (void)hipMemsetAsync(bins, 0, binsBytes, stream);  // 0 < ordenc(any real)
      k_scatter_max<<<(size_t)N * 128 / 256, 256, 0, stream>>>(netcat, idxb, bins, N, T);
      k_gather<<<(size_t)N * 128 / 256, 256, 0, stream>>>(netcat, idxb, bins, N, T);
    }
    const float* w0 = blk0_w + (size_t)i * 256 * 128;
    const float* b0 = blk0_b + (size_t)i * 128;
    const float* w1 = blk1_w + (size_t)i * 128 * 128;
    const float* b1 = blk1_b + (size_t)i * 128;
    const float* ws_ = blk_sc_w + (size_t)i * 256 * 128;
    // h = relu(x) @ w0 + b0
    k_gemm<256, true, true, false><<<gemmBlocks, 256, 0, stream>>>(
        netcat, 256, w0, b0, nullptr, 0, hbuf, 128);
    // t = relu(h) @ w1 + b1
    k_gemm<128, true, true, false><<<gemmBlocks, 256, 0, stream>>>(
        hbuf, 128, w1, b1, nullptr, 0, tbuf, 128);
    // net = x @ sc_w + t   (written into netcat cols 0..127; async-A path)
    k_gemm<256, false, false, true><<<gemmBlocks, 256, 0, stream>>>(
        netcat, 256, ws_, nullptr, tbuf, 128, netcat, 256);
  }

  // 4. c = net @ fc_c_w + b   (K=128: first 128 cols of netcat; async-A path)
  k_gemm<128, false, true, false><<<gemmBlocks, 256, 0, stream>>>(
      netcat, 256, fc_c_w, fc_c_b, nullptr, 0, hbuf, 128);

  // 5. scatter-mean into (3,B,128,64,64)
  const long totalOut = (long)3 * NBATCH * 128 * R2V;
  (void)hipMemsetAsync(out, 0, (size_t)totalOut * 4, stream);
  (void)hipMemsetAsync(cnt, 0, (size_t)3 * NBATCH * R2V * 4, stream);
  k_scatter_mean<<<(size_t)N * 128 / 256, 256, 0, stream>>>(hbuf, idxb, out, cnt, N, T);
  k_div<<<(totalOut + 255) / 256, 256, 0, stream>>>(out, cnt, totalOut);
}
